// Attention_46995532153259
// MI455X (gfx1250) — matrix operand order repaired and checked by reading, not compiled, not run on hardware
//
#include <hip/hip_runtime.h>
#include <hip/hip_bf16.h>

#define D_MODEL 1024
#define N_HEADS 16
#define D_HEAD  64
#define BATCH   2
#define SEQ     2048
#define MTILES  (SEQ / 32)   // 64 q-tiles of 32 rows

typedef _Float16 v16h __attribute__((ext_vector_type(16)));
typedef _Float16 v8h  __attribute__((ext_vector_type(8)));
typedef _Float16 v4h  __attribute__((ext_vector_type(4)));
typedef float    v8f  __attribute__((ext_vector_type(8)));
typedef float    f4   __attribute__((ext_vector_type(4)));

static __device__ __forceinline__ v8f wmma_f16(v16h a, v16h b, v8f c) {
    // D = A(16x32 f16) * B(32x16 f16) + C(16x16 f32)
    return __builtin_amdgcn_wmma_f32_16x16x32_f16(
        /*neg_a=*/false, a, /*neg_b=*/false, b,
        /*c_mod=*/(short)0, c, /*reuse_a=*/false, /*reuse_b=*/false);
}

// A-fragment (16x32 f16) from an f16 row-major row pointer.
// Lane holds row M = lane&15; element j -> K = k0+16*kh+j
static __device__ __forceinline__ v16h load_a_f16(const _Float16* rowp, int k0, int kh) {
    v8h lo = *(const v8h*)(rowp + k0 + 16 * kh);
    v8h hi = *(const v8h*)(rowp + k0 + 16 * kh + 8);
    v16h a;
#pragma unroll
    for (int j = 0; j < 8; ++j) { a[j] = lo[j]; a[j + 8] = hi[j]; }
    return a;
}

// ---------------------------------------------------------------------------
// Prep kernels (run once per launch; ~48 MB of traffic, L2-resident after)
// ---------------------------------------------------------------------------
__global__ void __launch_bounds__(256)
cvt_x_kernel(const float* __restrict__ x, _Float16* __restrict__ xh) {
    // B*S*D_MODEL = 4M elements, 4 per thread; packed 8B f16 store
    int i = (blockIdx.x * 256 + threadIdx.x) * 4;
    f4 v = *(const f4*)(x + i);
    v4h o;
    o[0] = (_Float16)v[0]; o[1] = (_Float16)v[1];
    o[2] = (_Float16)v[2]; o[3] = (_Float16)v[3];
    *(v4h*)(xh + i) = o;
}

__global__ void __launch_bounds__(256)
cvt_wqkv_kernel(const float* __restrict__ W, _Float16* __restrict__ Wt) {
    // W: [H, D_MODEL, D_HEAD] fp32 -> Wt: [H, D_HEAD, D_MODEL] f16 (K-major)
    int i = blockIdx.x * 256 + threadIdx.x;          // over H*D_MODEL*D_HEAD = 1M
    int e = i % D_HEAD;
    int t = i / D_HEAD;
    int d = t % D_MODEL;
    int h = t / D_MODEL;
    Wt[((size_t)h * D_HEAD + e) * D_MODEL + d] = (_Float16)W[i];
}

__global__ void __launch_bounds__(256)
cvt_wo_kernel(const float* __restrict__ W, _Float16* __restrict__ Wt) {
    // W_O [h,e,d] flattens to row-major [k=h*64+e, d]; store transposed [d, k]
    int i = blockIdx.x * 256 + threadIdx.x;          // over 1M
    int d = i % D_MODEL;
    int k = i / D_MODEL;
    Wt[(size_t)d * D_MODEL + k] = (_Float16)W[i];
}

// ---------------------------------------------------------------------------
// Kernel 1: per-head projection. One wave computes a 32(q)x64(dh) tile for one
// (b,h): two 16-row A sub-tiles share every B-fragment (K-major f16 weights),
// 8 WMMAs per 12 b128-loads in the K-loop.
// transposeOut==1 writes [b,h,dh,s] (for V), else [b,h,s,dh].
// ---------------------------------------------------------------------------
__global__ void __launch_bounds__(32)
qkv_proj_kernel(const _Float16* __restrict__ xh,   // [B,S,D_MODEL] f16
                const _Float16* __restrict__ Wt,   // [H,D_HEAD,D_MODEL] f16
                const float* __restrict__ bias,    // [H,D_HEAD] fp32
                _Float16* __restrict__ out,
                int transposeOut) {
    int id = blockIdx.x;
    int mt = id % MTILES; id /= MTILES;
    int h  = id % N_HEADS;
    int b  = id / N_HEADS;
    int lane = threadIdx.x;
    int m  = lane & 15;
    int kh = lane >> 4;
    int qbase = mt * 32;

    const _Float16* rowp0 = xh + ((size_t)b * SEQ + qbase + m) * D_MODEL;
    const _Float16* rowp1 = rowp0 + (size_t)16 * D_MODEL;
    const _Float16* wbase = Wt + (size_t)h * D_HEAD * D_MODEL;

    v8f acc[2][4];
#pragma unroll
    for (int qs = 0; qs < 2; ++qs)
#pragma unroll
        for (int nb = 0; nb < 4; ++nb)
#pragma unroll
            for (int i = 0; i < 8; ++i) acc[qs][nb][i] = 0.0f;

    for (int k0 = 0; k0 < D_MODEL; k0 += 32) {
        v16h a0 = load_a_f16(rowp0, k0, kh);
        v16h a1 = load_a_f16(rowp1, k0, kh);
#pragma unroll
        for (int nb = 0; nb < 4; ++nb) {
            int e = nb * 16 + m;                      // B col (N) = lane&15
            v16h bf = *(const v16h*)(wbase + (size_t)e * D_MODEL + k0 + 16 * kh);
            acc[0][nb] = wmma_f16(a0, bf, acc[0][nb]);
            acc[1][nb] = wmma_f16(a1, bf, acc[1][nb]);
        }
    }

#pragma unroll
    for (int qs = 0; qs < 2; ++qs)
#pragma unroll
        for (int nb = 0; nb < 4; ++nb) {
            int col = nb * 16 + m;
            float bv = bias[h * D_HEAD + col];
#pragma unroll
            for (int r = 0; r < 8; ++r) {
                int row = qbase + qs * 16 + r + 8 * kh;   // C vgpr r -> M=r+8*kh
                float v = acc[qs][nb][r] + bv;
                if (transposeOut)
                    out[(((size_t)b * N_HEADS + h) * D_HEAD + col) * SEQ + row] = (_Float16)v;
                else
                    out[(((size_t)b * N_HEADS + h) * SEQ + row) * D_HEAD + col] = (_Float16)v;
            }
        }
}

// ---------------------------------------------------------------------------
// Kernel 2: causal flash attention. One wave per (b,h,32-row q-tile); the two
// 16-row score/PV sub-tiles share all K and V fragments.
// Q,K: [b,h,s,dh] f16; Vt: [b,h,dh,s] f16; Z out: [b,s,h*dh] f16.
// ---------------------------------------------------------------------------
__global__ void __launch_bounds__(32)
attn_kernel(const _Float16* __restrict__ Q,
            const _Float16* __restrict__ K,
            const _Float16* __restrict__ Vt,
            _Float16* __restrict__ Z) {
    __shared__ _Float16 ldsP[2][16 * 32];   // P tiles: C-layout -> A-layout bounce

    int id = blockIdx.x;
    int mt = id % MTILES; id /= MTILES;
    int h  = id % N_HEADS;
    int b  = id / N_HEADS;
    int lane = threadIdx.x;
    int m  = lane & 15;
    int kh = lane >> 4;
    int qbase = mt * 32;
    size_t bh = (size_t)b * N_HEADS + h;

    // Q A-fragments: [q sub-tile][dh half], pre-scaled by 1/sqrt(dh)=0.125
    v16h aq[2][2];
#pragma unroll
    for (int qs = 0; qs < 2; ++qs) {
        const _Float16* qrow = Q + (bh * SEQ + qbase + qs * 16 + m) * D_HEAD;
        aq[qs][0] = load_a_f16(qrow, 0, kh);
        aq[qs][1] = load_a_f16(qrow, 32, kh);
#pragma unroll
        for (int j = 0; j < 16; ++j) {
            aq[qs][0][j] = aq[qs][0][j] * (_Float16)0.125f;
            aq[qs][1][j] = aq[qs][1][j] * (_Float16)0.125f;
        }
    }

    float mrow[2][8], lrow[2][8];
    v8f o[2][4];
#pragma unroll
    for (int qs = 0; qs < 2; ++qs) {
#pragma unroll
        for (int r = 0; r < 8; ++r) { mrow[qs][r] = -3.0e38f; lrow[qs][r] = 0.0f; }
#pragma unroll
        for (int nb = 0; nb < 4; ++nb)
#pragma unroll
            for (int i = 0; i < 8; ++i) o[qs][nb][i] = 0.0f;
    }

    int nkb = mt + 1;                    // 32-key blocks for causal coverage
    for (int kb = 0; kb < nkb; ++kb) {
        int kbase = kb * 32;

        // prefetch next block's K/V streams (global_prefetch_b8)
        if (kb + 1 < nkb) {
            __builtin_prefetch(K + (bh * SEQ + kbase + 32 + lane) * D_HEAD, 0, 1);
            __builtin_prefetch(Vt + (bh * D_HEAD + 2 * lane) * SEQ + kbase + 32, 0, 1);
        }

        // K B-fragments [key sub-tile t][dh half f], shared by both q sub-tiles
        v16h bk[2][2];
#pragma unroll
        for (int t = 0; t < 2; ++t) {
            const _Float16* kp = K + (bh * SEQ + kbase + t * 16 + m) * D_HEAD;
            bk[t][0] = *(const v16h*)(kp + 0  + 16 * kh);   // B: K(dh)=j+16*kh
            bk[t][1] = *(const v16h*)(kp + 32 + 16 * kh);
        }

        // scores: 4 C tiles (2 q sub-tiles x 2 key sub-tiles), 8 WMMAs
        v8f sc[2][2];
#pragma unroll
        for (int qs = 0; qs < 2; ++qs)
#pragma unroll
            for (int t = 0; t < 2; ++t) {
                v8f s;
#pragma unroll
                for (int i = 0; i < 8; ++i) s[i] = 0.0f;
                s = wmma_f16(aq[qs][0], bk[t][0], s);
                s = wmma_f16(aq[qs][1], bk[t][1], s);
                sc[qs][t] = s;
            }

        // causal mask + online softmax per q sub-tile
        __syncthreads();                 // protect LDS reads of previous block
#pragma unroll
        for (int qs = 0; qs < 2; ++qs) {
#pragma unroll
            for (int r = 0; r < 8; ++r) {
                int q  = qbase + qs * 16 + r + 8 * kh;
                int k0 = kbase + m;
                if (k0 > q)      sc[qs][0][r] = -1e5f;
                if (k0 + 16 > q) sc[qs][1][r] = -1e5f;
            }
#pragma unroll
            for (int r = 0; r < 8; ++r) {
                float vmax = fmaxf(sc[qs][0][r], sc[qs][1][r]);
#pragma unroll
                for (int off = 8; off >= 1; off >>= 1)
                    vmax = fmaxf(vmax, __shfl_xor(vmax, off, 32));
                float mnew  = fmaxf(mrow[qs][r], vmax);
                float alpha = __expf(mrow[qs][r] - mnew);
                float p0 = __expf(sc[qs][0][r] - mnew);
                float p1 = __expf(sc[qs][1][r] - mnew);
                float ps = p0 + p1;
#pragma unroll
                for (int off = 8; off >= 1; off >>= 1)
                    ps += __shfl_xor(ps, off, 32);
                lrow[qs][r] = lrow[qs][r] * alpha + ps;
                mrow[qs][r] = mnew;
#pragma unroll
                for (int nb = 0; nb < 4; ++nb) o[qs][nb][r] *= alpha;
                // stage P (f16) in LDS: row = r+8*kh, col = m (+16 for t=1)
                int row = r + 8 * kh;
                ldsP[qs][row * 32 + m]      = (_Float16)p0;
                ldsP[qs][row * 32 + 16 + m] = (_Float16)p1;
            }
        }
        __syncthreads();

        // read P back in A-layout
        v16h ap[2];
#pragma unroll
        for (int qs = 0; qs < 2; ++qs) {
            const _Float16* lp = &ldsP[qs][m * 32];
            v8h plo = *(const v8h*)(lp + 16 * kh);        // A half j<8  -> K=j+16*kh
            v8h phi = *(const v8h*)(lp + 16 * kh + 8);    // A half j>=8 -> K=j+16*kh
#pragma unroll
            for (int j = 0; j < 8; ++j) { ap[qs][j] = plo[j]; ap[qs][j + 8] = phi[j]; }
        }

        // PV: 4 V fragments shared by both q sub-tiles, 8 WMMAs
#pragma unroll
        for (int nb = 0; nb < 4; ++nb) {
            int dh = nb * 16 + m;
            const _Float16* vp = Vt + (bh * D_HEAD + dh) * SEQ + kbase + 16 * kh;
            v16h bv = *(const v16h*)vp;
            o[0][nb] = wmma_f16(ap[0], bv, o[0][nb]);
            o[1][nb] = wmma_f16(ap[1], bv, o[1][nb]);
        }
    }

    // normalize and write Z[b, q, h*64 + dh] (f16)
#pragma unroll
    for (int qs = 0; qs < 2; ++qs)
#pragma unroll
        for (int r = 0; r < 8; ++r) {
            float inv = 1.0f / lrow[qs][r];
            int q = qbase + qs * 16 + r + 8 * kh;
#pragma unroll
            for (int nb = 0; nb < 4; ++nb) {
                int col = h * D_HEAD + nb * 16 + m;
                Z[((size_t)b * SEQ + q) * D_MODEL + col] = (_Float16)(o[qs][nb][r] * inv);
            }
        }
}

// ---------------------------------------------------------------------------
// Kernel 3: output projection  out = Z[4096,1024] * Wo[1024,1024] + bO
// Wot pre-transposed f16 [d, k]; one wave per 32x64 output tile.
// ---------------------------------------------------------------------------
__global__ void __launch_bounds__(32)
oproj_kernel(const _Float16* __restrict__ Z,
             const _Float16* __restrict__ Wot,   // [D_MODEL(d), D_MODEL(k)] f16
             const float* __restrict__ bO,
             float* __restrict__ out) {
    const int NT = D_MODEL / 64;         // 16 col-tiles of 64
    int id = blockIdx.x;
    int nt = id % NT;
    int rt = id / NT;
    int lane = threadIdx.x;
    int m  = lane & 15;
    int kh = lane >> 4;
    int rowbase = rt * 32;
    int colbase = nt * 64;

    const _Float16* zrow0 = Z + ((size_t)rowbase + m) * D_MODEL;
    const _Float16* zrow1 = zrow0 + (size_t)16 * D_MODEL;

    v8f acc[2][4];
#pragma unroll
    for (int qs = 0; qs < 2; ++qs)
#pragma unroll
        for (int nb = 0; nb < 4; ++nb)
#pragma unroll
            for (int i = 0; i < 8; ++i) acc[qs][nb][i] = 0.0f;

    for (int k0 = 0; k0 < D_MODEL; k0 += 32) {
        v16h a0 = load_a_f16(zrow0, k0, kh);
        v16h a1 = load_a_f16(zrow1, k0, kh);
#pragma unroll
        for (int nb = 0; nb < 4; ++nb) {
            int d = colbase + nb * 16 + m;
            v16h bf = *(const v16h*)(Wot + (size_t)d * D_MODEL + k0 + 16 * kh);
            acc[0][nb] = wmma_f16(a0, bf, acc[0][nb]);
            acc[1][nb] = wmma_f16(a1, bf, acc[1][nb]);
        }
    }

#pragma unroll
    for (int qs = 0; qs < 2; ++qs)
#pragma unroll
        for (int nb = 0; nb < 4; ++nb) {
            int col = colbase + nb * 16 + m;
            float bv = bO[col];
#pragma unroll
            for (int r = 0; r < 8; ++r) {
                int row = rowbase + qs * 16 + r + 8 * kh;
                out[(size_t)row * D_MODEL + col] = acc[qs][nb][r] + bv;
            }
        }
}

// ---------------------------------------------------------------------------
extern "C" void kernel_launch(void* const* d_in, const int* in_sizes, int n_in,
                              void* d_out, int out_size, void* d_ws, size_t ws_size,
                              hipStream_t stream) {
    const float* x  = (const float*)d_in[0];
    const float* WQ = (const float*)d_in[1];
    const float* WK = (const float*)d_in[2];
    const float* WV = (const float*)d_in[3];
    const float* WO = (const float*)d_in[4];
    const float* bQ = (const float*)d_in[5];
    const float* bK = (const float*)d_in[6];
    const float* bV = (const float*)d_in[7];
    const float* bO = (const float*)d_in[8];
    float* out = (float*)d_out;

    // workspace layout (f16 elements):
    //   Qf | Kf | Vt | Zf : 4 x B*S*D_MODEL = 4 x 4M halves (32 MB)
    //   xh                : 4M halves (8 MB)
    //   Wqt | Wkt | Wvt   : 3 x 1M halves (6 MB)
    //   Wot               : 1M halves (2 MB)
    const size_t nper = (size_t)BATCH * SEQ * D_MODEL;       // 4M
    const size_t wqkv = (size_t)N_HEADS * D_HEAD * D_MODEL;  // 1M
    _Float16* Qf  = (_Float16*)d_ws;
    _Float16* Kf  = Qf  + nper;
    _Float16* Vt  = Kf  + nper;
    _Float16* Zf  = Vt  + nper;
    _Float16* xh  = Zf  + nper;
    _Float16* Wqt = xh  + nper;
    _Float16* Wkt = Wqt + wqkv;
    _Float16* Wvt = Wkt + wqkv;
    _Float16* Wot = Wvt + wqkv;

    dim3 b256(256);
    cvt_x_kernel   <<<(int)(nper / (256 * 4)), b256, 0, stream>>>(x, xh);
    cvt_wqkv_kernel<<<(int)(wqkv / 256), b256, 0, stream>>>(WQ, Wqt);
    cvt_wqkv_kernel<<<(int)(wqkv / 256), b256, 0, stream>>>(WK, Wkt);
    cvt_wqkv_kernel<<<(int)(wqkv / 256), b256, 0, stream>>>(WV, Wvt);
    cvt_wo_kernel  <<<(int)(wqkv / 256), b256, 0, stream>>>(WO, Wot);

    dim3 b32(32);
    int nblk = BATCH * N_HEADS * MTILES;            // 2048 waves
    qkv_proj_kernel<<<nblk, b32, 0, stream>>>(xh, Wqt, bQ, Qf, 0);
    qkv_proj_kernel<<<nblk, b32, 0, stream>>>(xh, Wkt, bK, Kf, 0);
    qkv_proj_kernel<<<nblk, b32, 0, stream>>>(xh, Wvt, bV, Vt, 1);
    attn_kernel<<<nblk, b32, 0, stream>>>(Qf, Kf, Vt, Zf);
    oproj_kernel<<<(BATCH * SEQ / 32) * (D_MODEL / 64), b32, 0, stream>>>(Zf, Wot, bO, out);
}
